// HeteroGraphNN_87660282511996
// MI455X (gfx1250) — compile-verified, matrix operand order checked by reading, not run on hardware
//
#include <hip/hip_runtime.h>
#include <hip/hip_bf16.h>

typedef __attribute__((ext_vector_type(16))) __bf16 v16bf;
typedef __attribute__((ext_vector_type(8)))  float  v8f;

#define D_FEAT 256
#define NU 50000
#define NI 50000
#define NE 320000
#define DRAW 1024

// relaxed agent-scope f32 atomic add -> global_atomic_add_f32
__device__ __forceinline__ void atomAddF(float* p, float v) {
    __hip_atomic_fetch_add(p, v, __ATOMIC_RELAXED, __HIP_MEMORY_SCOPE_AGENT);
}

// ---------------- pack f32 weights [K,N] row-major -> bf16 [K/32][N][32] ----------------
__global__ void pack_b_kernel(const float* __restrict__ W, __bf16* __restrict__ out,
                              int K, int N) {
    int idx = blockIdx.x * blockDim.x + threadIdx.x;
    if (idx >= K * N) return;
    int k = idx / N;
    int n = idx - k * N;
    out[(size_t)(k >> 5) * N * 32 + (size_t)n * 32 + (k & 31)] = (__bf16)W[idx];
}

// ---------------- zero (float4) ----------------
__global__ void zero_kernel(float4* __restrict__ p, long n4) {
    long i = (long)blockIdx.x * blockDim.x + threadIdx.x;
    if (i < n4) p[i] = make_float4(0.f, 0.f, 0.f, 0.f);
}

// ---------------- degree accumulation (one thread per edge) ----------------
__global__ void degree_kernel(const int* __restrict__ src, const int* __restrict__ dst,
                              float* __restrict__ dego, float* __restrict__ degi, int ne) {
    int e = blockIdx.x * blockDim.x + threadIdx.x;
    if (e >= ne) return;
    atomAddF(&dego[src[e]], 1.0f);
    atomAddF(&degi[dst[e]], 1.0f);
}

__global__ void rsqrt_kernel(float* __restrict__ p, int n) {
    int i = blockIdx.x * blockDim.x + threadIdx.x;
    if (i < n) p[i] = rsqrtf(fmaxf(p[i], 1.0f));
}

// ---------------- edge scatter: agg[dst] += x[src] * rs_out[src]  (one wave per edge) ----
__global__ void scatter_kernel(const float* __restrict__ x, const float* __restrict__ rso,
                               const int* __restrict__ src, const int* __restrict__ dst,
                               float* __restrict__ agg, int ne) {
    int wave = (int)((blockIdx.x * blockDim.x + threadIdx.x) >> 5);
    int lane = threadIdx.x & 31;
    if (wave >= ne) return;
    int s = src[wave];
    int d = dst[wave];
    float sc = rso[s];
    const float* xr = x + (size_t)s * D_FEAT;
    float* ar = agg + (size_t)d * D_FEAT;
#pragma unroll
    for (int i = 0; i < 8; ++i) {
        int f = lane + i * 32;
        atomAddF(&ar[f], xr[f] * sc);
    }
}

// ---------------- elementwise epilogues (float4) ----------------
__global__ void relu_res_kernel(float4* __restrict__ t, const float4* __restrict__ r, long n4) {
    long i = (long)blockIdx.x * blockDim.x + threadIdx.x;
    if (i >= n4) return;
    float4 a = t[i], b = r[i];
    a.x = fmaxf(a.x + b.x, 0.f); a.y = fmaxf(a.y + b.y, 0.f);
    a.z = fmaxf(a.z + b.z, 0.f); a.w = fmaxf(a.w + b.w, 0.f);
    t[i] = a;
}
__global__ void add_res_kernel(float4* __restrict__ t, const float4* __restrict__ r, long n4) {
    long i = (long)blockIdx.x * blockDim.x + threadIdx.x;
    if (i >= n4) return;
    float4 a = t[i], b = r[i];
    a.x += b.x; a.y += b.y; a.z += b.z; a.w += b.w;
    t[i] = a;
}

// ---------------- bf16 WMMA GEMM: C[M,N] (+)= (rowscale .* A[M,K]) @ B[K,N] + bias ------
// A: f32 row-major, converted to bf16 on the fly (optional per-row scale fused in).
// Bp: bf16 packed [K/32][N][32] -> per-lane fragments are one 32B contiguous load.
// 8 waves / block; each wave computes a 16x128 strip, double-buffered K pipeline.
// Requires M % 16 == 0, N % 128 == 0, K % 64 == 0.
__global__ __launch_bounds__(256)
void gemm_bf16_kernel(const float* __restrict__ A, const float* __restrict__ rowscale,
                      const __bf16* __restrict__ Bp, const float* __restrict__ bias,
                      float* __restrict__ C, int M, int N, int K, int accum) {
    int wave = threadIdx.x >> 5;
    int lane = threadIdx.x & 31;
    int mTile = blockIdx.x * 8 + wave;
    if (mTile * 16 >= M) return;
    const int mBase   = mTile * 16;
    const int colBase = blockIdx.y * 128;
    const int halfsel = lane >> 4;   // 0: K 0-15 / M 0-7 ; 1: K 16-31 / M 8-15
    const int lan     = lane & 15;

    v8f c[8];
#pragma unroll
    for (int j = 0; j < 8; ++j)
#pragma unroll
        for (int v = 0; v < 8; ++v) c[j][v] = 0.0f;

    const int arow = mBase + lan;                 // M % 16 == 0 -> always in range
    const float* Arow = A + (size_t)arow * K;
    const float sc = rowscale ? rowscale[arow] : 1.0f;

    // lane-invariant-per-loop base: B fragment address = BkBase + (k0/32)*panel + j*512
    const __bf16* BkBase = Bp + (size_t)colBase * 32 + (size_t)lan * 32 + halfsel * 16;
    const size_t panelStride = (size_t)N * 32;

    auto LOAD_A = [&](int kk, float4 (&dst)[4]) {
        const float* p = Arow + kk + halfsel * 16;
        dst[0] = *(const float4*)(p + 0);
        dst[1] = *(const float4*)(p + 4);
        dst[2] = *(const float4*)(p + 8);
        dst[3] = *(const float4*)(p + 12);
    };
    auto LOAD_B = [&](int kk, v16bf (&dst)[8]) {
        const __bf16* p = BkBase + (size_t)(kk >> 5) * panelStride;
#pragma unroll
        for (int j = 0; j < 8; ++j) dst[j] = *(const v16bf*)(p + j * 512);
    };
    auto COMPUTE = [&](const float4 (&as)[4], const v16bf (&b)[8]) {
        float af[16] = {as[0].x, as[0].y, as[0].z, as[0].w,
                        as[1].x, as[1].y, as[1].z, as[1].w,
                        as[2].x, as[2].y, as[2].z, as[2].w,
                        as[3].x, as[3].y, as[3].z, as[3].w};
        v16bf a;
#pragma unroll
        for (int t = 0; t < 16; ++t) a[t] = (__bf16)(af[t] * sc);
#pragma unroll
        for (int j = 0; j < 8; ++j)
            c[j] = __builtin_amdgcn_wmma_f32_16x16x32_bf16(
                false, a, false, b[j], (short)0, c[j], false, false);
    };

    // double-buffered pipeline, K % 64 == 0
    float4 aA[4], aB[4];
    v16bf  bA[8], bB[8];
    LOAD_A(0, aA); LOAD_B(0, bA);
    for (int k0 = 0; k0 < K; k0 += 64) {
        LOAD_A(k0 + 32, aB); LOAD_B(k0 + 32, bB);   // prefetch stage 1
        COMPUTE(aA, bA);                            // compute stage 0
        if (k0 + 64 < K) { LOAD_A(k0 + 64, aA); LOAD_B(k0 + 64, bA); }  // prefetch stage 2
        COMPUTE(aB, bB);                            // compute stage 1
    }

    // ---- epilogue: bias add, optional accumulate ----
    const int rbase = mBase + halfsel * 8;
#pragma unroll
    for (int j = 0; j < 8; ++j) {
        int col = colBase + j * 16 + lan;
        float bv = bias[col];
#pragma unroll
        for (int v = 0; v < 8; ++v) {
            size_t idx = (size_t)(rbase + v) * N + col;
            float val = c[j][v] + bv;
            if (accum) val += C[idx];
            C[idx] = val;
        }
    }
}

extern "C" void kernel_launch(void* const* d_in, const int* in_sizes, int n_in,
                              void* d_out, int out_size, void* d_ws, size_t ws_size,
                              hipStream_t stream) {
    const float* x_user = (const float*)d_in[0];
    const float* x_item = (const float*)d_in[1];
    const int* srcF  = (const int*)d_in[2];
    const int* dstF  = (const int*)d_in[3];
    const int* srcR  = (const int*)d_in[4];
    const int* dstR  = (const int*)d_in[5];
    const int* srcRB = (const int*)d_in[6];
    const int* dstRB = (const int*)d_in[7];
    const float* Wr = (const float*)d_in[8];
    const float* br = (const float*)d_in[9];
    const float* W1 = (const float*)d_in[10];
    const float* b1 = (const float*)d_in[11];
    const float* W2 = (const float*)d_in[12];
    const float* b2 = (const float*)d_in[13];
    const float* Wi = (const float*)d_in[14];
    const float* bi = (const float*)d_in[15];
    float* out = (float*)d_out;

    // ---- workspace layout ----
    const size_t F = (size_t)NU * D_FEAT;       // 12.8M floats per node-feature buffer
    float* wsf   = (float*)d_ws;
    float* bufU0 = wsf;                         // u0, later conv2 user output / g_u
    float* bufI0 = wsf + F;                     // i0, later conv2 item output / g_i
    float* bufTU = wsf + 2 * F;                 // conv1 user accum -> h_u
    float* bufTI = wsf + 3 * F;                 // conv1 item accum -> h_i
    float* aggU  = wsf + 4 * F;                 // scatter target (user dst)
    float* aggI  = wsf + 5 * F;                 // scatter target (item dst)
    float* degs  = wsf + 6 * F;                 // 6 x 50000 rsqrt-degree arrays
    float* rs_fo_o = degs;                      // follows: out-deg (user)
    float* rs_fo_i = degs + 50000;              // follows: in-deg  (user)
    float* rs_ra_o = degs + 100000;             // rates:   out-deg (user)
    float* rs_ra_i = degs + 150000;             // rates:   in-deg  (item)
    float* rs_rb_o = degs + 200000;             // ratedby: out-deg (item)
    float* rs_rb_i = degs + 250000;             // ratedby: in-deg  (user)
    __bf16* packs = (__bf16*)(degs + 300000);   // packed bf16 weights
    __bf16* Wr_p  = packs;                      // 1024*256
    __bf16* W1_p  = packs + 262144;             // 3 * 256*256
    __bf16* W2_p  = packs + 262144 + 196608;
    __bf16* Wi_p  = packs + 262144 + 2 * 196608; // 256*1024

    // ---- pack weights to bf16 fragment layout ----
    pack_b_kernel<<<1024, 256, 0, stream>>>(Wr, Wr_p, 1024, 256);
    for (int r = 0; r < 3; ++r) {
        pack_b_kernel<<<256, 256, 0, stream>>>(W1 + (size_t)r * 65536, W1_p + (size_t)r * 65536, 256, 256);
        pack_b_kernel<<<256, 256, 0, stream>>>(W2 + (size_t)r * 65536, W2_p + (size_t)r * 65536, 256, 256);
    }
    pack_b_kernel<<<1024, 256, 0, stream>>>(Wi, Wi_p, 256, 1024);

    // ---- degrees -> rsqrt(max(deg,1)) ----
    zero_kernel<<<(300000 / 4 + 255) / 256, 256, 0, stream>>>((float4*)degs, 300000 / 4);
    degree_kernel<<<(NE + 255) / 256, 256, 0, stream>>>(srcF, dstF, rs_fo_o, rs_fo_i, NE);
    degree_kernel<<<(NE + 255) / 256, 256, 0, stream>>>(srcR, dstR, rs_ra_o, rs_ra_i, NE);
    degree_kernel<<<(NE + 255) / 256, 256, 0, stream>>>(srcRB, dstRB, rs_rb_o, rs_rb_i, NE);
    rsqrt_kernel<<<(300000 + 255) / 256, 256, 0, stream>>>(degs, 300000);

    const dim3 blk(256);
    const dim3 gemm_g256(391, 2);     // M=50000 (3125 row-tiles / 8 waves), N=256
    const dim3 gemm_g1024(391, 8);    // N=1024
    const long n4 = (long)(F / 4);
    const int v4Blocks  = (int)((n4 + 255) / 256);
    const int scatBlocks = NE / 8;    // one wave per edge, 8 waves per block

    // ---- dim_reduce: u0 = x_user @ Wr + br ; i0 = x_item @ Wr + br ----
    gemm_bf16_kernel<<<gemm_g256, blk, 0, stream>>>(x_user, nullptr, Wr_p, br, bufU0, NU, 256, 1024, 0);
    gemm_bf16_kernel<<<gemm_g256, blk, 0, stream>>>(x_item, nullptr, Wr_p, br, bufI0, NI, 256, 1024, 0);

    // ---- conv1, user dst: gconv(follows, u0, W1[0]) + gconv(ratedby, i0, W1[2]) ----
    zero_kernel<<<v4Blocks, blk, 0, stream>>>((float4*)aggU, n4);
    scatter_kernel<<<scatBlocks, blk, 0, stream>>>(bufU0, rs_fo_o, srcF, dstF, aggU, NE);
    gemm_bf16_kernel<<<gemm_g256, blk, 0, stream>>>(aggU, rs_fo_i, W1_p, b1, bufTU, NU, 256, 256, 0);
    zero_kernel<<<v4Blocks, blk, 0, stream>>>((float4*)aggU, n4);
    scatter_kernel<<<scatBlocks, blk, 0, stream>>>(bufI0, rs_rb_o, srcRB, dstRB, aggU, NE);
    gemm_bf16_kernel<<<gemm_g256, blk, 0, stream>>>(aggU, rs_rb_i, W1_p + 2 * 65536, b1 + 512, bufTU, NU, 256, 256, 1);
    // ---- conv1, item dst: gconv(rates, u0, W1[1]) ----
    zero_kernel<<<v4Blocks, blk, 0, stream>>>((float4*)aggI, n4);
    scatter_kernel<<<scatBlocks, blk, 0, stream>>>(bufU0, rs_ra_o, srcR, dstR, aggI, NE);
    gemm_bf16_kernel<<<gemm_g256, blk, 0, stream>>>(aggI, rs_ra_i, W1_p + 65536, b1 + 256, bufTI, NI, 256, 256, 0);
    // ---- h = relu(conv1 + x0) ----
    relu_res_kernel<<<v4Blocks, blk, 0, stream>>>((float4*)bufTU, (const float4*)bufU0, n4);
    relu_res_kernel<<<v4Blocks, blk, 0, stream>>>((float4*)bufTI, (const float4*)bufI0, n4);

    // ---- conv2, user dst (output into bufU0, residual added after) ----
    zero_kernel<<<v4Blocks, blk, 0, stream>>>((float4*)aggU, n4);
    scatter_kernel<<<scatBlocks, blk, 0, stream>>>(bufTU, rs_fo_o, srcF, dstF, aggU, NE);
    gemm_bf16_kernel<<<gemm_g256, blk, 0, stream>>>(aggU, rs_fo_i, W2_p, b2, bufU0, NU, 256, 256, 0);
    zero_kernel<<<v4Blocks, blk, 0, stream>>>((float4*)aggU, n4);
    scatter_kernel<<<scatBlocks, blk, 0, stream>>>(bufTI, rs_rb_o, srcRB, dstRB, aggU, NE);
    gemm_bf16_kernel<<<gemm_g256, blk, 0, stream>>>(aggU, rs_rb_i, W2_p + 2 * 65536, b2 + 512, bufU0, NU, 256, 256, 1);
    // ---- conv2, item dst ----
    zero_kernel<<<v4Blocks, blk, 0, stream>>>((float4*)aggI, n4);
    scatter_kernel<<<scatBlocks, blk, 0, stream>>>(bufTU, rs_ra_o, srcR, dstR, aggI, NE);
    gemm_bf16_kernel<<<gemm_g256, blk, 0, stream>>>(aggI, rs_ra_i, W2_p + 65536, b2 + 256, bufI0, NI, 256, 256, 0);
    // ---- g = conv2 + h ----
    add_res_kernel<<<v4Blocks, blk, 0, stream>>>((float4*)bufU0, (const float4*)bufTU, n4);
    add_res_kernel<<<v4Blocks, blk, 0, stream>>>((float4*)bufI0, (const float4*)bufTI, n4);

    // ---- dim_increase: out[0] = g_u @ Wi + bi ; out[1] = g_i @ Wi + bi ----
    gemm_bf16_kernel<<<gemm_g1024, blk, 0, stream>>>(bufU0, nullptr, Wi_p, bi, out, NU, 1024, 256, 0);
    gemm_bf16_kernel<<<gemm_g1024, blk, 0, stream>>>(bufI0, nullptr, Wi_p, bi, out + (size_t)NU * 1024, NI, 1024, 256, 0);
}
